// MLACausalSelfAttention_89232240542342
// MI455X (gfx1250) — matrix operand order repaired and checked by reading, not compiled, not run on hardware
//
#include <hip/hip_runtime.h>
#include <hip/hip_bf16.h>

typedef __attribute__((ext_vector_type(16))) _Float16 v16h;
typedef __attribute__((ext_vector_type(8)))  float    v8f;
typedef unsigned int u32x4 __attribute__((ext_vector_type(4)));
typedef unsigned int tdm_g0_t __attribute__((ext_vector_type(4)));
typedef int          tdm_g1_t __attribute__((ext_vector_type(8)));
typedef int          tdm_g2_t __attribute__((ext_vector_type(4)));
typedef int          tdm_g4_t __attribute__((ext_vector_type(8)));

constexpr int NB  = 2;
constexpr int NT  = 2048;
constexpr int NC  = 2048;
constexpr int NH  = 16;
constexpr int NKH = 4;
constexpr int NHD = 128;
constexpr int NDL = 512;
constexpr float EPSF   = 1.1920929e-07f;
constexpr float QSCALE = 0.08838834764831845f; // 1/sqrt(128)

// ---------------------------------------------------------------------------
// helpers
// ---------------------------------------------------------------------------
static __device__ __forceinline__ v16h ld_frag2(const _Float16* p0, const _Float16* p1) {
  union { v16h v; u32x4 q[2]; } u;
  u.q[0] = *(const u32x4*)p0;
  u.q[1] = *(const u32x4*)p1;
  return u.v;
}

// CDNA5 async memory->LDS copy, 16B per lane, tracked by ASYNCcnt.
static __device__ __forceinline__ void async_b128(_Float16* lds_dst, const _Float16* g_src) {
  unsigned lds = (unsigned)(unsigned long long)(size_t)lds_dst;
  unsigned long long ga = (unsigned long long)(size_t)g_src;
  asm volatile("global_load_async_to_lds_b128 %0, %1, off"
               :: "v"(lds), "v"(ga)
               : "memory");
}
static __device__ __forceinline__ void wait_async0() {
  asm volatile("s_wait_asynccnt 0x0" ::: "memory");
}

// CDNA5 Tensor Data Mover: one DMA for a 2D tile of 128 rows x 32 halves,
// with LDS padding 16 DW data + 4 DW pad -> 80B pitch (= ASP*2).
// D# group0: count=1 | lds_addr | global_addr | type=2
// D# group1: data_size=2B, pad_enable, pad_interval=16DW, pad_amount=4DW,
//            tensor_dim0=row_len, tensor_dim1=big, tile_dim0=32, tile_dim1=128,
//            tensor_dim0_stride=row_stride
static __device__ __forceinline__ void tdm_load_tile(
    unsigned lds_byte_addr, const _Float16* gaddr, int row_stride_elems)
{
  unsigned long long ga = (unsigned long long)(size_t)gaddr;
  unsigned td0 = (unsigned)row_stride_elems;   // dim-0 length (elements)
  unsigned td1 = 1u << 20;                     // plenty of rows
  unsigned str = (unsigned)row_stride_elems;   // dim-0 stride (elements)

  tdm_g0_t g0;
  g0[0] = 1u;                                            // count=1, user desc
  g0[1] = lds_byte_addr;                                 // lds_addr
  g0[2] = (unsigned)(ga & 0xFFFFFFFFull);                // global_addr[31:0]
  g0[3] = (unsigned)((ga >> 32) & 0x01FFFFFFull) | 0x80000000u; // ga[56:32] | type=2

  tdm_g1_t g1;
  g1[0] = (1 << 16) | (1 << 20) | (3 << 22) | (3 << 25); // data_size=1(2B), pad_en, intv=16DW, amt=4DW
  g1[1] = (int)((td0 & 0xFFFFu) << 16);                  // abar=0 | tensor_dim0[15:0]
  g1[2] = (int)(((td0 >> 16) & 0xFFFFu) | ((td1 & 0xFFFFu) << 16)); // td0[31:16] | td1[15:0]
  g1[3] = (int)(((td1 >> 16) & 0xFFFFu) | (32u << 16));  // td1[31:16] | tile_dim0=32
  g1[4] = (int)128u;                                     // tile_dim1=128, tile_dim2=0
  g1[5] = (int)str;                                      // tensor_dim0_stride[31:0]
  g1[6] = 0;                                             // stride[47:32] | td1_stride[15:0]
  g1[7] = 0;

  tdm_g2_t g2 = {0, 0, 0, 0};
  tdm_g2_t g3 = {0, 0, 0, 0};
  tdm_g4_t g4 = {0, 0, 0, 0, 0, 0, 0, 0};
  __builtin_amdgcn_tensor_load_to_lds(g0, g1, g2, g3, g4, 0);
}

// ---------------------------------------------------------------------------
// f32 -> f16 cast (grid-stride)
// ---------------------------------------------------------------------------
__global__ void cast_f16_kernel(const float* __restrict__ in, _Float16* __restrict__ out, size_t n) {
  size_t i = (size_t)blockIdx.x * blockDim.x + threadIdx.x;
  size_t stride = (size_t)gridDim.x * blockDim.x;
  for (; i < n; i += stride) out[i] = (_Float16)in[i];
}

// f32 (K x N row-major) -> f16 transposed (N x K row-major)
__global__ void cast_transpose_f16_kernel(const float* __restrict__ in, _Float16* __restrict__ out,
                                          int Kdim, int Ndim) {
  size_t n = (size_t)Kdim * Ndim;
  size_t i = (size_t)blockIdx.x * blockDim.x + threadIdx.x;
  size_t stride = (size_t)gridDim.x * blockDim.x;
  for (; i < n; i += stride) {
    int k = (int)(i / Ndim);
    int c = (int)(i % Ndim);
    out[(size_t)c * Kdim + k] = (_Float16)in[i];
  }
}

// ---------------------------------------------------------------------------
// WMMA GEMM: C[f32] = A[f16, MxK row-major] * Bt^T, with Bt = [N x K] f16
// Tile 128x128x32, 256 threads = 8 waves, wave = 64x32 of C.
// Double-buffered LDS: A tile via global_load_async_to_lds_b128 (per lane),
// B tile via one TDM tensor_load_to_lds per k-step (wave 0).
// grid.x = N/128, grid.y = M/128
// ---------------------------------------------------------------------------
#define GBM 128
#define GBK 32
#define ASP 40   // LDS pitch in halves (TDM pad reproduces this: 64B data + 16B pad)

__global__ __launch_bounds__(256) void gemm_f16_wmma(
    const _Float16* __restrict__ A, const _Float16* __restrict__ Bt,
    float* __restrict__ C, int K, int lda, int ldb, int ldc)
{
  __shared__ _Float16 As[2][GBM * ASP];
  __shared__ _Float16 Bs[2][GBM * ASP];   // [n][k]

  const int tid  = threadIdx.x;
  const int wid  = tid >> 5;
  const int lane = tid & 31;
  const int l15  = lane & 15;
  const int lhi  = lane >> 4;

  const int row0 = blockIdx.y * GBM;
  const int col0 = blockIdx.x * GBM;
  const int wm   = wid >> 2;   // 0..1 -> 64 rows
  const int wn   = wid & 3;    // 0..3 -> 32 cols

  const int r   = (tid * 2) >> 2;       // staging row
  const int sg0 = (tid * 2) & 3;
  const int sg1 = sg0 + 1;

  auto stage = [&](int buf, int k0) {
    // A tile: 128 rows x 32 halves, per-lane async b128
    async_b128(&As[buf][r * ASP + sg0 * 8], A + (size_t)(row0 + r) * lda + k0 + sg0 * 8);
    async_b128(&As[buf][r * ASP + sg1 * 8], A + (size_t)(row0 + r) * lda + k0 + sg1 * 8);
    // B tile: one TDM DMA (128 rows x 64B, padded to 80B pitch in LDS)
    if (wid == 0) {
      tdm_load_tile((unsigned)(size_t)&Bs[buf][0],
                    Bt + (size_t)col0 * ldb + k0, ldb);
    }
  };
  auto wait_stage = [&]() {
    wait_async0();
    if (wid == 0) __builtin_amdgcn_s_wait_tensorcnt(0);
    __syncthreads();
  };

  v8f acc[4][2] = {};

  stage(0, 0);
  wait_stage();

  int cur = 0;
  for (int k0 = 0; k0 < K; k0 += GBK) {
    const int nxt = cur ^ 1;
    if (k0 + GBK < K) stage(nxt, k0 + GBK);
    if (k0 + 2 * GBK < K) {
      __builtin_prefetch(A + (size_t)(row0 + r) * lda + k0 + 2 * GBK, 0, 1);
      __builtin_prefetch(Bt + (size_t)(col0 + r) * ldb + k0 + 2 * GBK, 0, 1);
    }

    // ---- fragments from LDS (buffer `cur`)
    v16h af[4], bf[2];
#pragma unroll
    for (int mt = 0; mt < 4; ++mt) {
      const _Float16* p = &As[cur][(wm * 64 + mt * 16 + l15) * ASP];
      af[mt] = ld_frag2(p + lhi * 8, p + 16 + lhi * 8);
    }
#pragma unroll
    for (int nt = 0; nt < 2; ++nt) {
      const _Float16* p = &Bs[cur][(wn * 32 + nt * 16 + l15) * ASP + lhi * 8];
      bf[nt] = ld_frag2(p, p + 16);
    }
#pragma unroll
    for (int mt = 0; mt < 4; ++mt)
#pragma unroll
      for (int nt = 0; nt < 2; ++nt)
        acc[mt][nt] = __builtin_amdgcn_wmma_f32_16x16x32_f16(
            false, af[mt], false, bf[nt], (short)0, acc[mt][nt], false, false);

    wait_stage();
    cur = nxt;
  }

  // ---- epilogue (C layout: VGPR r -> M = r + 8*lhi, N = l15)
#pragma unroll
  for (int mt = 0; mt < 4; ++mt)
#pragma unroll
    for (int nt = 0; nt < 2; ++nt) {
      int mbase = row0 + wm * 64 + mt * 16 + lhi * 8;
      int n = col0 + wn * 32 + nt * 16 + l15;
#pragma unroll
      for (int rr = 0; rr < 8; ++rr)
        C[(size_t)(mbase + rr) * ldc + n] = acc[mt][nt][rr];
    }
}

// ---------------------------------------------------------------------------
// build q: rope first 64 dims, rms over 128, fold 1/sqrt(HD), write f16
// ---------------------------------------------------------------------------
__global__ __launch_bounds__(256) void build_q_kernel(
    const float* __restrict__ qf, const float* __restrict__ cosb,
    const float* __restrict__ sinb, _Float16* __restrict__ qh)
{
  int row  = blockIdx.x * 8 + (threadIdx.x >> 5);  // (b*NT+t)*NH + h
  int lane = threadIdx.x & 31;
  size_t bt = (size_t)row / NH;
  int h  = row % NH;
  int t  = (int)(bt % NT);
  const float* src = qf + bt * (size_t)(NH * NHD) + h * NHD;
  float c = cosb[t * 64 + lane];
  float s = sinb[t * 64 + lane];
  float e0 = src[lane], e1 = src[lane + 32], e2 = src[lane + 64], e3 = src[lane + 96];
  float r0 = e0 * c + e1 * s;
  float r1 = -e0 * s + e1 * c;
  float ss = r0 * r0 + r1 * r1 + e2 * e2 + e3 * e3;
#pragma unroll
  for (int m = 1; m < 32; m <<= 1) ss += __shfl_xor(ss, m, 32);
  float inv = rsqrtf(ss * (1.0f / 128.0f) + EPSF) * QSCALE;
  _Float16* dst = qh + (size_t)row * NHD;
  dst[lane]      = (_Float16)(r0 * inv);
  dst[lane + 32] = (_Float16)(r1 * inv);
  dst[lane + 64] = (_Float16)(e2 * inv);
  dst[lane + 96] = (_Float16)(e3 * inv);
}

// ---------------------------------------------------------------------------
// build k: k = [rope(k_rope) | k_nope], rms over 128, write f16
// ---------------------------------------------------------------------------
__global__ __launch_bounds__(256) void build_k_kernel(
    const float* __restrict__ kropef, const float* __restrict__ knopef,
    const float* __restrict__ cosb, const float* __restrict__ sinb,
    _Float16* __restrict__ kh)
{
  int row  = blockIdx.x * 8 + (threadIdx.x >> 5);  // (b*NT+t)*NKH + head
  int lane = threadIdx.x & 31;
  size_t bt = (size_t)row / NKH;
  int head = row % NKH;
  int t    = (int)(bt % NT);
  const float* kr = kropef + bt * (size_t)(NKH * 64) + head * 64;
  const float* kn = knopef + bt * (size_t)(NKH * 64) + head * 64;
  float c = cosb[t * 64 + lane];
  float s = sinb[t * 64 + lane];
  float e0 = kr[lane], e1 = kr[lane + 32];
  float e2 = kn[lane], e3 = kn[lane + 32];
  float r0 = e0 * c + e1 * s;
  float r1 = -e0 * s + e1 * c;
  float ss = r0 * r0 + r1 * r1 + e2 * e2 + e3 * e3;
#pragma unroll
  for (int m = 1; m < 32; m <<= 1) ss += __shfl_xor(ss, m, 32);
  float inv = rsqrtf(ss * (1.0f / 128.0f) + EPSF);
  _Float16* dst = kh + (size_t)row * NHD;
  dst[lane]      = (_Float16)(r0 * inv);
  dst[lane + 32] = (_Float16)(r1 * inv);
  dst[lane + 64] = (_Float16)(e2 * inv);
  dst[lane + 96] = (_Float16)(e3 * inv);
}

// ---------------------------------------------------------------------------
// v (b,t,kvh,d) f32 -> vT (b,kvh,d,t) f16
// ---------------------------------------------------------------------------
__global__ void build_vT_kernel(const float* __restrict__ v, _Float16* __restrict__ vT, size_t n) {
  size_t i = (size_t)blockIdx.x * blockDim.x + threadIdx.x;
  size_t stride = (size_t)gridDim.x * blockDim.x;
  for (; i < n; i += stride) {
    int d = (int)(i % NHD);
    size_t r = i / NHD;
    int h = (int)(r % NKH);
    size_t bt = r / NKH;
    int t = (int)(bt % NT);
    int b = (int)(bt / NT);
    vT[(((size_t)b * NKH + h) * NHD + d) * NT + t] = (_Float16)v[i];
  }
}

// ---------------------------------------------------------------------------
// Flash attention, causal, GQA rep=4.
// grid = (T/128, H, B), block = 256 (8 waves). Wave = 16 q rows.
// ---------------------------------------------------------------------------
__global__ __launch_bounds__(256) void attn_kernel(
    const _Float16* __restrict__ qh,  // (B,T,H,128), pre-scaled
    const _Float16* __restrict__ kh,  // (B,T,KH,128)
    const _Float16* __restrict__ vT,  // (B,KH,128,T)
    float* __restrict__ y)            // (B,T,H,128)
{
  __shared__ _Float16 Plds[8][16 * 32 + 8];

  const int tid  = threadIdx.x;
  const int wid  = tid >> 5;
  const int lane = tid & 31;
  const int l15  = lane & 15;
  const int lhi  = lane >> 4;

  const int h   = blockIdx.y;
  const int b   = blockIdx.z;
  const int qb  = blockIdx.x * 128 + wid * 16;  // wave's first q token
  const int kvh = h >> 2;

  v16h qa[4];
#pragma unroll
  for (int kd = 0; kd < 4; ++kd) {
    const _Float16* p = qh + (((size_t)b * NT + qb + l15) * NH + h) * NHD + kd * 32;
    qa[kd] = ld_frag2(p + lhi * 8, p + 16 + lhi * 8);
  }

  v8f o[8] = {};
  float m_r[8], l_r[8];
#pragma unroll
  for (int r = 0; r < 8; ++r) { m_r[r] = -3.0e38f; l_r[r] = 0.0f; }

  _Float16* pl = Plds[wid];
  const int lastc = (qb + 15) >> 5;

  for (int c = 0; c <= lastc; ++c) {
    const int c0 = c * 32;

    v8f s[2] = {};
#pragma unroll
    for (int nt = 0; nt < 2; ++nt) {
      const _Float16* kbase =
          kh + (((size_t)b * NT + c0 + nt * 16 + l15) * NKH + kvh) * NHD + lhi * 8;
#pragma unroll
      for (int kd = 0; kd < 4; ++kd) {
        v16h bf = ld_frag2(kbase + kd * 32, kbase + kd * 32 + 16);
        s[nt] = __builtin_amdgcn_wmma_f32_16x16x32_f16(
            false, qa[kd], false, bf, (short)0, s[nt], false, false);
      }
    }

    if (c0 + 31 > qb) {
#pragma unroll
      for (int nt = 0; nt < 2; ++nt) {
        int kc = c0 + nt * 16 + l15;
#pragma unroll
        for (int r = 0; r < 8; ++r) {
          int qr = qb + lhi * 8 + r;
          if (kc > qr) s[nt][r] = -3.0e38f;
        }
      }
    }

    float mnew[8];
#pragma unroll
    for (int r = 0; r < 8; ++r) {
      float v = fmaxf(s[0][r], s[1][r]);
      v = fmaxf(v, __shfl_xor(v, 1, 32));
      v = fmaxf(v, __shfl_xor(v, 2, 32));
      v = fmaxf(v, __shfl_xor(v, 4, 32));
      v = fmaxf(v, __shfl_xor(v, 8, 32));
      mnew[r] = fmaxf(m_r[r], v);
    }

#pragma unroll
    for (int nt = 0; nt < 2; ++nt)
#pragma unroll
      for (int r = 0; r < 8; ++r) {
        float p = __expf(s[nt][r] - mnew[r]);
        s[nt][r] = p;
        pl[(lhi * 8 + r) * 32 + nt * 16 + l15] = (_Float16)p;
      }

#pragma unroll
    for (int r = 0; r < 8; ++r) {
      float ps = s[0][r] + s[1][r];
      ps += __shfl_xor(ps, 1, 32);
      ps += __shfl_xor(ps, 2, 32);
      ps += __shfl_xor(ps, 4, 32);
      ps += __shfl_xor(ps, 8, 32);
      float corr = __expf(m_r[r] - mnew[r]);
      l_r[r] = l_r[r] * corr + ps;
      m_r[r] = mnew[r];
#pragma unroll
      for (int d = 0; d < 8; ++d) o[d][r] *= corr;
    }

    v16h pa = ld_frag2(pl + l15 * 32 + lhi * 8, pl + l15 * 32 + 16 + lhi * 8);

#pragma unroll
    for (int d = 0; d < 8; ++d) {
      const _Float16* vp =
          vT + (((size_t)b * NKH + kvh) * NHD + d * 16 + l15) * NT + c0 + lhi * 8;
      v16h bf = ld_frag2(vp, vp + 16);
      o[d] = __builtin_amdgcn_wmma_f32_16x16x32_f16(
          false, pa, false, bf, (short)0, o[d], false, false);
    }
  }

#pragma unroll
  for (int d = 0; d < 8; ++d) {
#pragma unroll
    for (int r = 0; r < 8; ++r) {
      int qr = qb + lhi * 8 + r;
      y[(((size_t)b * NT + qr) * NH + h) * NHD + d * 16 + l15] = o[d][r] / l_r[r];
    }
  }
}

// ---------------------------------------------------------------------------
// launch
// ---------------------------------------------------------------------------
extern "C" void kernel_launch(void* const* d_in, const int* in_sizes, int n_in,
                              void* d_out, int out_size, void* d_ws, size_t ws_size,
                              hipStream_t stream) {
  const float* x      = (const float*)d_in[0];
  const float* cosb   = (const float*)d_in[1];
  const float* sinb   = (const float*)d_in[2];
  const float* Wq     = (const float*)d_in[3];
  const float* Wkv    = (const float*)d_in[4];
  const float* Wkup   = (const float*)d_in[5];
  const float* Wvup   = (const float*)d_in[6];
  const float* Wkrope = (const float*)d_in[7];
  const float* Wproj  = (const float*)d_in[8];
  float* out = (float*)d_out;

  char* ws = (char*)d_ws;
  size_t off = 0;
  auto alloc = [&](size_t bytes) -> void* {
    void* p = ws + off;
    off += (bytes + 255) & ~(size_t)255;
    return p;
  };

  const size_t nBT = (size_t)NB * NT;                 // 4096
  _Float16* xh      = (_Float16*)alloc(nBT * NC * 2);
  _Float16* WqT     = (_Float16*)alloc((size_t)NC * NC * 2);     // [N=2048][K=2048]
  _Float16* WkvT    = (_Float16*)alloc((size_t)NDL * NC * 2);    // [512][2048]
  _Float16* WkupT   = (_Float16*)alloc((size_t)256 * NDL * 2);   // [256][512]
  _Float16* WvupT   = (_Float16*)alloc((size_t)512 * NDL * 2);   // [512][512]
  _Float16* WkrT    = (_Float16*)alloc((size_t)256 * NC * 2);    // [256][2048]
  _Float16* WpjT    = (_Float16*)alloc((size_t)NC * NC * 2);     // [2048][2048]
  float*    qf      = (float*)alloc(nBT * 2048 * 4);
  float*    latf    = (float*)alloc(nBT * NDL * 4);
  _Float16* lat_h   = (_Float16*)alloc(nBT * NDL * 2);
  float*    knopef  = (float*)alloc(nBT * 256 * 4);
  float*    kropef  = (float*)alloc(nBT * 256 * 4);
  float*    vf      = (float*)alloc(nBT * 512 * 4);
  _Float16* qh      = (_Float16*)alloc(nBT * 2048 * 2);
  _Float16* khb     = (_Float16*)alloc(nBT * 512 * 2);
  _Float16* vT      = (_Float16*)alloc(nBT * 512 * 2);
  float*    yf      = (float*)alloc(nBT * 2048 * 4);
  _Float16* yh      = (_Float16*)alloc(nBT * 2048 * 2);
  (void)ws_size; (void)in_sizes; (void)n_in; (void)out_size;

  // ---- casts / weight transposes (one-time, bandwidth-trivial)
  cast_f16_kernel<<<2048, 256, 0, stream>>>(x, xh, nBT * NC);
  cast_transpose_f16_kernel<<<2048, 256, 0, stream>>>(Wq, WqT, NC, NC);
  cast_transpose_f16_kernel<<<1024, 256, 0, stream>>>(Wkv, WkvT, NC, NDL);
  cast_transpose_f16_kernel<<<512,  256, 0, stream>>>(Wkup, WkupT, NDL, 256);
  cast_transpose_f16_kernel<<<512,  256, 0, stream>>>(Wvup, WvupT, NDL, 512);
  cast_transpose_f16_kernel<<<1024, 256, 0, stream>>>(Wkrope, WkrT, NC, 256);
  cast_transpose_f16_kernel<<<2048, 256, 0, stream>>>(Wproj, WpjT, NC, NC);

  // ---- projection GEMMs (grid = N/128 x M/128); ldb = K of the transposed weight
  gemm_f16_wmma<<<dim3(2048 / 128, 4096 / 128), 256, 0, stream>>>(
      xh, WqT, qf, 2048, 2048, 2048, 2048);
  gemm_f16_wmma<<<dim3(512 / 128, 4096 / 128), 256, 0, stream>>>(
      xh, WkvT, latf, 2048, 2048, 2048, 512);
  gemm_f16_wmma<<<dim3(256 / 128, 4096 / 128), 256, 0, stream>>>(
      xh, WkrT, kropef, 2048, 2048, 2048, 256);
  cast_f16_kernel<<<2048, 256, 0, stream>>>(latf, lat_h, nBT * NDL);
  gemm_f16_wmma<<<dim3(256 / 128, 4096 / 128), 256, 0, stream>>>(
      lat_h, WkupT, knopef, 512, 512, 512, 256);
  gemm_f16_wmma<<<dim3(512 / 128, 4096 / 128), 256, 0, stream>>>(
      lat_h, WvupT, vf, 512, 512, 512, 512);

  // ---- rope + rms + casts/transposes
  build_q_kernel<<<(int)(nBT * NH / 8), 256, 0, stream>>>(qf, cosb, sinb, qh);
  build_k_kernel<<<(int)(nBT * NKH / 8), 256, 0, stream>>>(kropef, knopef, cosb, sinb, khb);
  build_vT_kernel<<<2048, 256, 0, stream>>>(vf, vT, nBT * 512);

  // ---- flash attention
  attn_kernel<<<dim3(NT / 128, NH, NB), 256, 0, stream>>>(qh, khb, vT, yf);

  // ---- output projection
  cast_f16_kernel<<<2048, 256, 0, stream>>>(yf, yh, nBT * 2048);
  gemm_f16_wmma<<<dim3(2048 / 128, 4096 / 128), 256, 0, stream>>>(
      yh, WpjT, out, 2048, 2048, 2048, 2048);
}
